// _CausalSelfAttentionRelativePosition_2869038154196
// MI455X (gfx1250) — matrix operand order repaired and checked by reading, not compiled, not run on hardware
//
#include <hip/hip_runtime.h>
#include <hip/hip_bf16.h>

typedef unsigned short u16t;
typedef unsigned int   u32t;
typedef unsigned long long u64t;

typedef __attribute__((ext_vector_type(16))) __bf16 v16bf;
typedef __attribute__((ext_vector_type(8)))  float  v8f;
typedef __attribute__((ext_vector_type(8)))  u32t   v8u;

#define B_  4
#define L_  2048
#define D_  1024
#define H_  16
#define HS_ 64

#define WMMA_BF16(a, b, c) \
  __builtin_amdgcn_wmma_f32_16x16x32_bf16(false, (a), false, (b), (short)0, (c), false, false)

static __device__ __forceinline__ u16t f2bf(float f) {
  u32t u = __float_as_uint(f);
  u32t r = (u + 0x7FFFu + ((u >> 16) & 1u)) >> 16;  // round-to-nearest-even
  return (u16t)r;
}

// ---- WMMA A fragment (16x32 bf16) from row-major source ----------------
// ISA layout: lanes 0-15 row M=lane; VGPR0..3: K=0..7, VGPR4..7: K=16..23;
// lanes 16-31 same rows, K offset +8.
static __device__ __forceinline__ v16bf load_a(const u16t* base, int ld, int row0, int koff) {
  const int lane = threadIdx.x & 31;
  const int m = lane & 15, hv = lane >> 4;
  const u16t* p = base + (size_t)(row0 + m) * ld + koff + hv * 8;
  v8u d;
#pragma unroll
  for (int v = 0; v < 4; ++v) d[v] = *(const u32t*)(p + 2 * v);
#pragma unroll
  for (int v = 0; v < 4; ++v) d[4 + v] = *(const u32t*)(p + 16 + 2 * v);
  return __builtin_bit_cast(v16bf, d);
}

// ---- WMMA B fragment (32x16 bf16) from B^T row-major storage -----------
// Lane holds column N=lane&15; lanes 0-15 carry K=0..7,16..23, lanes 16-31 K=8..15,24..31
// (same K order as the A fragment).
static __device__ __forceinline__ v16bf load_b(const u16t* baseT, int ld, int col0, int koff) {
  const int lane = threadIdx.x & 31;
  const int n = lane & 15, hv = lane >> 4;
  const u16t* p = baseT + (size_t)(col0 + n) * ld + koff + hv * 8;
  v8u d;
#pragma unroll
  for (int v = 0; v < 8; ++v) d[v] = *(const u32t*)(p + (v >> 2) * 8 + 2 * v);
  return __builtin_bit_cast(v16bf, d);
}

static __device__ __forceinline__ v16bf load_b_clamped(const u16t* baseT, int ld, int col0,
                                                       int maxcol, int koff) {
  const int lane = threadIdx.x & 31;
  const int n = lane & 15, hv = lane >> 4;
  int c = col0 + n;
  c = c < 0 ? 0 : (c > maxcol ? maxcol : c);
  const u16t* p = baseT + (size_t)c * ld + koff + hv * 8;
  v8u d;
#pragma unroll
  for (int v = 0; v < 8; ++v) d[v] = *(const u32t*)(p + (v >> 2) * 8 + 2 * v);
  return __builtin_bit_cast(v16bf, d);
}

// ---- async stage of one 32(N)x32(K) bf16 B^T tile into LDS -------------
// 256 threads x 8 bytes = 2 KB. ASYNCcnt-tracked (no VGPR round trip).
// LDS byte address = low 32 bits of the generic pointer (flat aperture).
static __device__ __forceinline__ void stage_b_tile(u16t* ldsDst /*[32][32]*/,
                                                    const u16t* srcT, int ld,
                                                    int col0, int ks) {
  const int tid = threadIdx.x;           // 0..255
  const int n = tid >> 3, kp = (tid & 7) * 4;   // 4 bf16 = 8 bytes per lane
  const u16t* gp = srcT + (size_t)(col0 + n) * ld + ks + kp;
  const u32t lds_addr = (u32t)(uintptr_t)(ldsDst + n * 32 + kp);
  asm volatile("global_load_async_to_lds_b64 %0, %1, off"
               :: "v"(lds_addr), "v"((u64t)(uintptr_t)gp)
               : "memory");
}
static __device__ __forceinline__ void wait_async_le1() {
  asm volatile("s_wait_asynccnt 0x1" ::: "memory");
}
static __device__ __forceinline__ void wait_async_0() {
  asm volatile("s_wait_asynccnt 0x0" ::: "memory");
}

// ======================= K0: precision/transpose prep ===================
__global__ void __launch_bounds__(256) prep_kernel(
    const float* __restrict__ x, const float* __restrict__ Wa,
    const float* __restrict__ Wp, const float* __restrict__ Er,
    u16t* __restrict__ xbf, u16t* __restrict__ wta,
    u16t* __restrict__ wtp, u16t* __restrict__ erb) {
  const int NX = B_ * L_ * D_;
  const int NWA = D_ * 3 * D_;
  const int NWP = D_ * D_;
  const int NER = L_ * HS_;
  const int total = NX + NWA + NWP + NER;
  for (int i = blockIdx.x * blockDim.x + threadIdx.x; i < total;
       i += gridDim.x * blockDim.x) {
    if (i < NX) {
      xbf[i] = f2bf(x[i]);
    } else if (i < NX + NWA) {
      int j = i - NX; int n = j >> 10, k = j & 1023;           // wta[n][k] = Wa[k][n]
      wta[j] = f2bf(Wa[(size_t)k * (3 * D_) + n]);
    } else if (i < NX + NWA + NWP) {
      int j = i - NX - NWA; int n = j >> 10, k = j & 1023;     // wtp[n][k] = Wp[k][n]
      wtp[j] = f2bf(Wp[(size_t)k * D_ + n]);
    } else {
      int j = i - NX - NWA - NWP;
      erb[j] = f2bf(Er[j]);
    }
  }
}

// ======================= K1: QKV GEMM + head split ======================
// Block = 8 waves computing a 128(M)x32(N) macro tile; the shared 32x32 B^T
// tile is double-buffered in LDS via async-to-LDS loads.
__global__ void __launch_bounds__(256) qkv_kernel(
    const u16t* __restrict__ xbf, const u16t* __restrict__ wta,
    const float* __restrict__ bias,
    u16t* __restrict__ qb, u16t* __restrict__ kbuf, u16t* __restrict__ vt) {
  __shared__ u16t ldsB[2][32][32];
  const int NT = (3 * D_) / 32;                 // 96 col tiles
  const int w = threadIdx.x >> 5;
  const int row0 = (blockIdx.x / NT) * 128 + w * 16;
  const int col0 = (blockIdx.x % NT) * 32;
  const int lane = threadIdx.x & 31, n = lane & 15, hv = lane >> 4;

  v8f C0 = {}, C1 = {};
  stage_b_tile(&ldsB[0][0][0], wta, D_, col0, 0);
  const int NS = D_ / 32;                       // 32 k-steps
  for (int s = 0; s < NS; ++s) {
    if (s + 1 < NS) {
      stage_b_tile(&ldsB[(s + 1) & 1][0][0], wta, D_, col0, (s + 1) * 32);
      wait_async_le1();                         // stage s complete
    } else {
      wait_async_0();
    }
    __syncthreads();
    v16bf a  = load_a(xbf, D_, row0, s * 32);
    v16bf b0 = load_b(&ldsB[s & 1][0][0], 32, 0, 0);
    v16bf b1 = load_b(&ldsB[s & 1][0][0], 32, 16, 0);
    C0 = WMMA_BF16(a, b0, C0);
    C1 = WMMA_BF16(a, b1, C1);
    __syncthreads();                            // tile consumed by all waves
  }
#pragma unroll
  for (int r = 0; r < 8; ++r) {
    const int M = r + 8 * hv;
    const int row = row0 + M;
    const int bb = row >> 11, tt = row & (L_ - 1);
#pragma unroll
    for (int t = 0; t < 2; ++t) {
      const int col = col0 + 16 * t + n;
      const float v = (t == 0 ? C0[r] : C1[r]) + bias[col];
      const int sec = col >> 10;                 // 0=q 1=k 2=v
      const int d = col & (D_ - 1);
      const int h = d >> 6, hs = d & 63;
      const size_t head = (size_t)(bb * H_ + h);
      const u16t bv = f2bf(v);
      if (sec == 0)      qb  [(head * L_ + tt) * HS_ + hs] = bv;
      else if (sec == 1) kbuf[(head * L_ + tt) * HS_ + hs] = bv;
      else               vt  [(head * HS_ + hs) * L_ + tt] = bv;
    }
  }
}

// ======================= K2: flash attention + rel-pos ==================
// Srel[q,k] = q . Er[L-1+k-q]  (music-transformer skew, start==0).
// One wave per (b,h, 16-query tile); keys streamed in blocks of 32 so the
// softmax P tile (16x32) is exactly one WMMA A fragment for P@V.
__global__ void __launch_bounds__(128) attn_kernel(
    const u16t* __restrict__ qb, const u16t* __restrict__ kbuf,
    const u16t* __restrict__ vt, const u16t* __restrict__ erb,
    u16t* __restrict__ yb) {
  __shared__ float ldsR[4][16][48];   // relative-band scores, per wave slot
  __shared__ u16t  ldsP[4][16][32];   // P tile staging (C-layout -> A-layout)

  const int w    = threadIdx.x >> 5;
  const int lane = threadIdx.x & 31;
  const int n    = lane & 15;
  const int hv   = lane >> 4;
  const int wave = blockIdx.x * 4 + w;
  const int qt = wave & (L_ / 16 - 1);
  const int bh = wave >> 7;
  const int q0 = qt * 16;

  const u16t* qh = qb   + (size_t)bh * L_ * HS_;
  const u16t* kh = kbuf + (size_t)bh * L_ * HS_;
  const u16t* vh = vt   + (size_t)bh * HS_ * L_;

  const v16bf qa0 = load_a(qh, HS_, q0, 0);
  const v16bf qa1 = load_a(qh, HS_, q0, 32);

  v8f O0 = {}, O1 = {}, O2 = {}, O3 = {};
  float rm[8], rl[8];
#pragma unroll
  for (int r = 0; r < 8; ++r) { rm[r] = -1e30f; rl[r] = 0.f; }

  const int nkb = (q0 + 16 + 31) >> 5;
  for (int kbI = 0; kbI < nkb; ++kbI) {
    const int k0 = kbI << 5;

    // ---- S = Q K^T, 16x32 ----
    v8f S0 = {}, S1 = {};
    {
      v16bf b00 = load_b(kh, HS_, k0, 0);
      v16bf b01 = load_b(kh, HS_, k0, 32);
      S0 = WMMA_BF16(qa0, b00, S0);
      S0 = WMMA_BF16(qa1, b01, S0);
      v16bf b10 = load_b(kh, HS_, k0 + 16, 0);
      v16bf b11 = load_b(kh, HS_, k0 + 16, 32);
      S1 = WMMA_BF16(qa0, b10, S1);
      S1 = WMMA_BF16(qa1, b11, S1);
    }

    // ---- relative band R[qi][j] = q . Er[m0+j], j=0..47 ----
    const int m0 = (L_ - 1) + k0 - q0 - 15;
#pragma unroll
    for (int t = 0; t < 3; ++t) {
      v8f R = {};
      v16bf rb0 = load_b_clamped(erb, HS_, m0 + 16 * t, L_ - 1, 0);
      v16bf rb1 = load_b_clamped(erb, HS_, m0 + 16 * t, L_ - 1, 32);
      R = WMMA_BF16(qa0, rb0, R);
      R = WMMA_BF16(qa1, rb1, R);
#pragma unroll
      for (int r = 0; r < 8; ++r)
        ldsR[w][r + 8 * hv][16 * t + n] = R[r];
    }
    asm volatile("" ::: "memory");  // DS ops are in-order per wave

    // ---- scores: add rel, scale, causal mask ----
    float s0[8], s1[8];
    const float scale = 0.125f;     // 1/sqrt(64)
#pragma unroll
    for (int r = 0; r < 8; ++r) {
      const int M = r + 8 * hv;
      const int gq = q0 + M;
      const int j0 = n + 15 - M;    // 0..30
      float a = (S0[r] + ldsR[w][M][j0]) * scale;
      if (k0 + n > gq) a = -1e30f;
      s0[r] = a;
      float b = (S1[r] + ldsR[w][M][j0 + 16]) * scale;
      if (k0 + 16 + n > gq) b = -1e30f;
      s1[r] = b;
    }

    // ---- online softmax: row max across 16 lanes of each half ----
    float corr[8];
#pragma unroll
    for (int r = 0; r < 8; ++r) {
      float v = fmaxf(s0[r], s1[r]);
      v = fmaxf(v, __shfl_xor(v, 1, 32));
      v = fmaxf(v, __shfl_xor(v, 2, 32));
      v = fmaxf(v, __shfl_xor(v, 4, 32));
      v = fmaxf(v, __shfl_xor(v, 8, 32));
      const float mnew = fmaxf(rm[r], v);
      corr[r] = __expf(rm[r] - mnew);
      rm[r] = mnew;
    }

    // ---- P = exp(s - m): stage bf16 to LDS, accumulate row sums ----
#pragma unroll
    for (int r = 0; r < 8; ++r) {
      const int M = r + 8 * hv;
      const float p0 = __expf(s0[r] - rm[r]);
      const float p1 = __expf(s1[r] - rm[r]);
      ldsP[w][M][n]      = f2bf(p0);
      ldsP[w][M][n + 16] = f2bf(p1);
      float ps = p0 + p1;
      ps += __shfl_xor(ps, 1, 32);
      ps += __shfl_xor(ps, 2, 32);
      ps += __shfl_xor(ps, 4, 32);
      ps += __shfl_xor(ps, 8, 32);
      rl[r] = rl[r] * corr[r] + ps;
    }
#pragma unroll
    for (int r = 0; r < 8; ++r) {
      O0[r] *= corr[r]; O1[r] *= corr[r]; O2[r] *= corr[r]; O3[r] *= corr[r];
    }
    asm volatile("" ::: "memory");

    // ---- O += P @ V ----
    const v16bf pa = load_a(&ldsP[w][0][0], 32, 0, 0);
    O0 = WMMA_BF16(pa, load_b(vh, L_, 0,  k0), O0);
    O1 = WMMA_BF16(pa, load_b(vh, L_, 16, k0), O1);
    O2 = WMMA_BF16(pa, load_b(vh, L_, 32, k0), O2);
    O3 = WMMA_BF16(pa, load_b(vh, L_, 48, k0), O3);
  }

  // ---- normalize + store y as bf16 [B*L, D] ----
  const int bidx = bh >> 4, h = bh & 15;
#pragma unroll
  for (int r = 0; r < 8; ++r) {
    const int M = r + 8 * hv;
    const size_t row = (size_t)(bidx * L_ + q0 + M) * D_ + h * HS_;
    const float inv = 1.0f / rl[r];
    yb[row + n]      = f2bf(O0[r] * inv);
    yb[row + 16 + n] = f2bf(O1[r] * inv);
    yb[row + 32 + n] = f2bf(O2[r] * inv);
    yb[row + 48 + n] = f2bf(O3[r] * inv);
  }
}

// ======================= K3: output projection ==========================
__global__ void __launch_bounds__(256) proj_kernel(
    const u16t* __restrict__ ybf, const u16t* __restrict__ wtp,
    const float* __restrict__ bias, float* __restrict__ out) {
  __shared__ u16t ldsB[2][32][32];
  const int NT = D_ / 32;                       // 32 col tiles
  const int w = threadIdx.x >> 5;
  const int row0 = (blockIdx.x / NT) * 128 + w * 16;
  const int col0 = (blockIdx.x % NT) * 32;
  const int lane = threadIdx.x & 31, n = lane & 15, hv = lane >> 4;

  v8f C0 = {}, C1 = {};
  stage_b_tile(&ldsB[0][0][0], wtp, D_, col0, 0);
  const int NS = D_ / 32;
  for (int s = 0; s < NS; ++s) {
    if (s + 1 < NS) {
      stage_b_tile(&ldsB[(s + 1) & 1][0][0], wtp, D_, col0, (s + 1) * 32);
      wait_async_le1();
    } else {
      wait_async_0();
    }
    __syncthreads();
    v16bf a  = load_a(ybf, D_, row0, s * 32);
    v16bf b0 = load_b(&ldsB[s & 1][0][0], 32, 0, 0);
    v16bf b1 = load_b(&ldsB[s & 1][0][0], 32, 16, 0);
    C0 = WMMA_BF16(a, b0, C0);
    C1 = WMMA_BF16(a, b1, C1);
    __syncthreads();
  }
#pragma unroll
  for (int r = 0; r < 8; ++r) {
    const size_t row = (size_t)(row0 + r + 8 * hv) * D_;
    out[row + col0 + n]      = C0[r] + bias[col0 + n];
    out[row + col0 + 16 + n] = C1[r] + bias[col0 + 16 + n];
  }
}

// ======================= launch =========================================
extern "C" void kernel_launch(void* const* d_in, const int* in_sizes, int n_in,
                              void* d_out, int out_size, void* d_ws, size_t ws_size,
                              hipStream_t stream) {
  (void)in_sizes; (void)n_in; (void)out_size; (void)ws_size;
  const float* x  = (const float*)d_in[0];
  const float* Wa = (const float*)d_in[1];
  const float* ba = (const float*)d_in[2];
  const float* Wp = (const float*)d_in[3];
  const float* bp = (const float*)d_in[4];
  const float* Er = (const float*)d_in[5];
  float* out = (float*)d_out;

  char* ws = (char*)d_ws;
  size_t off = 0;
  auto wsalloc = [&](size_t bytes) -> char* {
    off = (off + 255) & ~(size_t)255;
    char* p = ws + off;
    off += bytes;
    return p;
  };
  u16t* xbf = (u16t*)wsalloc((size_t)B_ * L_ * D_ * 2);
  u16t* wta = (u16t*)wsalloc((size_t)3 * D_ * D_ * 2);
  u16t* wtp = (u16t*)wsalloc((size_t)D_ * D_ * 2);
  u16t* erb = (u16t*)wsalloc((size_t)L_ * HS_ * 2);
  u16t* qb  = (u16t*)wsalloc((size_t)B_ * H_ * L_ * HS_ * 2);
  u16t* kb  = (u16t*)wsalloc((size_t)B_ * H_ * L_ * HS_ * 2);
  u16t* vt  = (u16t*)wsalloc((size_t)B_ * H_ * HS_ * L_ * 2);
  u16t* yb  = (u16t*)wsalloc((size_t)B_ * L_ * D_ * 2);

  prep_kernel<<<4096, 256, 0, stream>>>(x, Wa, Wp, Er, xbf, wta, wtp, erb);

  const int qkv_blocks = ((B_ * L_) / 128) * ((3 * D_) / 32);   // 64*96 = 6144
  qkv_kernel<<<qkv_blocks, 256, 0, stream>>>(xbf, wta, ba, qb, kb, vt);

  const int attn_blocks = (B_ * H_ * (L_ / 16)) / 4;            // 2048
  attn_kernel<<<attn_blocks, 128, 0, stream>>>(qb, kb, vt, erb, yb);

  const int proj_blocks = ((B_ * L_) / 128) * (D_ / 32);        // 64*32 = 2048
  proj_kernel<<<proj_blocks, 256, 0, stream>>>(yb, wtp, bp, out);
}
